// GraphConvolution_50792283242910
// MI455X (gfx1250) — matrix operand order repaired and checked by reading, not compiled, not run on hardware
//
#include <hip/hip_runtime.h>
#include <hip/hip_bf16.h>

typedef __attribute__((ext_vector_type(16))) __bf16 v16bf;
typedef __attribute__((ext_vector_type(8)))  float  v8f;

constexpr int ND   = 128;   // node feature dim D
constexpr int EDIM = 16;    // edge feature dim DE
constexpr int HD   = 128;   // hidden H
constexpr int MD   = 128;   // message dim DM
constexpr int KXR  = 272;   // 2*ND + EDIM
constexpr int KX   = 288;   // KXR padded to multiple of 32
constexpr int KN   = 256;   // ND + MD (node MLP fan_in)
constexpr int TIL  = 32;    // rows (edges / nodes) per block (2 M-tiles)

// Native HW bf16 convert (RNE, NaN-correct) instead of manual bit tricks.
__device__ __forceinline__ unsigned short f2bf(float f) {
    union { __bf16 h; unsigned short s; } u;
    u.h = (__bf16)f;
    return u.s;
}

__device__ __forceinline__ unsigned int pack2bf(float lo, float hi) {
    return (unsigned int)f2bf(lo) | ((unsigned int)f2bf(hi) << 16);
}

union AF { v16bf v; unsigned int u[8]; };

// Build one A fragment (16x32 bf16) from an LDS row of packed bf16.
__device__ __forceinline__ AF loadA(const unsigned short* row, int k0, int hf) {
    AF a;
    #pragma unroll
    for (int r = 0; r < 4; ++r) {
        a.u[r]     = *(const unsigned int*)&row[k0 + 8*hf + 2*r];
        a.u[4 + r] = *(const unsigned int*)&row[k0 + 16 + 8*hf + 2*r];
    }
    return a;
}

// Build one B fragment from the pre-packed LDS tile WsT[128][16] (dword = K pair).
__device__ __forceinline__ AF loadB(const unsigned int (*WsT)[16], int col, int hf) {
    AF b;
    #pragma unroll
    for (int r = 0; r < 4; ++r) { b.u[r] = WsT[col][4*hf + r]; b.u[4 + r] = WsT[col][8 + 4*hf + r]; }   // -> 2x ds_load_b128
    return b;
}

// ---------------------------------------------------------------------------
// Edge kernel: message = silu(x@We1+be1)@We2+be2 ; gated ; atomic scatter-add
// 32 edges per block, 8 waves; each wave owns 16 output cols x 2 row tiles.
// ---------------------------------------------------------------------------
__global__ __launch_bounds__(256) void gnn_edge_kernel(
    const float* __restrict__ nf, const int* __restrict__ ei,
    const float* __restrict__ ef,
    const float* __restrict__ We1, const float* __restrict__ be1,
    const float* __restrict__ We2, const float* __restrict__ be2,
    const float* __restrict__ Wi,  const float* __restrict__ bi,
    float* __restrict__ agg, int nE)
{
    __shared__ __align__(16) unsigned short Xs[TIL][KX];   // 32x288 bf16
    __shared__ __align__(16) unsigned int   WsT[HD][16];   // B tile, packed K-pairs
    __shared__ __align__(16) unsigned short Hs[TIL][HD];   // 32x128 bf16
    __shared__ float Msg[TIL][MD];
    __shared__ float gates[TIL];
    __shared__ int   sSrc[TIL], sDst[TIL];

    const int t   = threadIdx.x;
    const int e0  = blockIdx.x * TIL;
    const int wv  = t >> 5;
    const int ln  = t & 31;
    const int hf  = ln >> 4;
    const int lm  = ln & 15;
    const int col = wv * 16 + lm;

    if (t < TIL)        sSrc[t]       = ei[e0 + t];
    else if (t < 2*TIL) sDst[t - TIL] = ei[nE + e0 + (t - TIL)];
    __syncthreads();

    // gather + concat -> bf16 LDS tile (zero-padded K 272..287)
    for (int i = t; i < TIL * KX; i += 256) {
        int m = i / KX, k = i - m * KX;
        float v;
        if (k < ND)          v = nf[(long)sSrc[m] * ND + k];
        else if (k < 2 * ND) v = nf[(long)sDst[m] * ND + (k - ND)];
        else if (k < KXR)    v = ef[(long)(e0 + m) * EDIM + (k - 2 * ND)];
        else                 v = 0.0f;
        Xs[m][k] = f2bf(v);
    }
    __syncthreads();

    // ---- matmul 1: [32 x 288] @ [288 x 128] ----
    v8f acc0 = {0.f,0.f,0.f,0.f,0.f,0.f,0.f,0.f};
    v8f acc1 = acc0;
    for (int k0 = 0; k0 < KX; k0 += 32) {
        for (int i = t; i < 16 * HD; i += 256) {            // stage packed B tile
            int kp = i >> 7, c = i & 127;
            int kg = k0 + 2 * kp;
            float v0 = (kg     < KXR) ? We1[kg * HD + c]       : 0.0f;
            float v1 = (kg + 1 < KXR) ? We1[(kg + 1) * HD + c] : 0.0f;
            WsT[c][kp] = pack2bf(v0, v1);
        }
        __syncthreads();
        AF b  = loadB(WsT, col, hf);
        AF a0 = loadA(Xs[lm],      k0, hf);
        AF a1 = loadA(Xs[16 + lm], k0, hf);
        acc0 = __builtin_amdgcn_wmma_f32_16x16x32_bf16(false, a0.v, false, b.v,
                                                       (short)0, acc0, false, false);
        acc1 = __builtin_amdgcn_wmma_f32_16x16x32_bf16(false, a1.v, false, b.v,
                                                       (short)0, acc1, false, false);
        __syncthreads();
    }

    {   // bias + SiLU -> Hs
        float bb = be1[col];
        #pragma unroll
        for (int r = 0; r < 8; ++r) {
            float x = acc0[r] + bb;
            Hs[r + 8*hf][col] = f2bf(x / (1.0f + __expf(-x)));
            float y = acc1[r] + bb;
            Hs[16 + r + 8*hf][col] = f2bf(y / (1.0f + __expf(-y)));
        }
    }
    __syncthreads();

    // ---- matmul 2: [32 x 128] @ [128 x 128] ----
    acc0 = (v8f){0.f,0.f,0.f,0.f,0.f,0.f,0.f,0.f};
    acc1 = acc0;
    for (int k0 = 0; k0 < MD; k0 += 32) {
        for (int i = t; i < 16 * HD; i += 256) {
            int kp = i >> 7, c = i & 127;
            int kg = k0 + 2 * kp;
            WsT[c][kp] = pack2bf(We2[kg * MD + c], We2[(kg + 1) * MD + c]);
        }
        __syncthreads();
        AF b  = loadB(WsT, col, hf);
        AF a0 = loadA(Hs[lm],      k0, hf);
        AF a1 = loadA(Hs[16 + lm], k0, hf);
        acc0 = __builtin_amdgcn_wmma_f32_16x16x32_bf16(false, a0.v, false, b.v,
                                                       (short)0, acc0, false, false);
        acc1 = __builtin_amdgcn_wmma_f32_16x16x32_bf16(false, a1.v, false, b.v,
                                                       (short)0, acc1, false, false);
        __syncthreads();
    }

    {   // message (pre-gate, f32) -> LDS
        float bb = be2[col];
        #pragma unroll
        for (int r = 0; r < 8; ++r) {
            Msg[r + 8*hf][col]      = acc0[r] + bb;
            Msg[16 + r + 8*hf][col] = acc1[r] + bb;
        }
    }
    __syncthreads();

    if (t < TIL) {       // sigmoid gate per edge row
        float g = bi[0];
        for (int c = 0; c < MD; ++c) g += Msg[t][c] * Wi[c];
        gates[t] = 1.0f / (1.0f + __expf(-g));
    }
    __syncthreads();

    // gated scatter-add into per-source-node aggregate (L2-resident)
    for (int i = t; i < TIL * MD; i += 256) {
        int m = i >> 7, c = i & 127;
        atomicAdd(&agg[(long)sSrc[m] * MD + c], Msg[m][c] * gates[m]);
    }
}

// ---------------------------------------------------------------------------
// Node kernel: out = nf + silu([nf|agg]@Wn1+bn1)@Wn2+bn2  (32 nodes / block)
// ---------------------------------------------------------------------------
__global__ __launch_bounds__(256) void gnn_node_kernel(
    const float* __restrict__ nf,  const float* __restrict__ agg,
    const float* __restrict__ Wn1, const float* __restrict__ bn1,
    const float* __restrict__ Wn2, const float* __restrict__ bn2,
    float* __restrict__ out, int N)
{
    __shared__ __align__(16) unsigned short Xs[TIL][KN];
    __shared__ __align__(16) unsigned int   WsT[HD][16];
    __shared__ __align__(16) unsigned short Hs[TIL][HD];

    const int t   = threadIdx.x;
    const int n0  = blockIdx.x * TIL;
    const int wv  = t >> 5;
    const int ln  = t & 31;
    const int hf  = ln >> 4;
    const int lm  = ln & 15;
    const int col = wv * 16 + lm;

    for (int i = t; i < TIL * KN; i += 256) {
        int m = i >> 8, k = i & 255;
        long row = n0 + m; if (row >= N) row = N - 1;       // clamp (guarded store)
        float v = (k < ND) ? nf[row * ND + k] : agg[row * MD + (k - ND)];
        Xs[m][k] = f2bf(v);
    }
    __syncthreads();

    v8f acc0 = {0.f,0.f,0.f,0.f,0.f,0.f,0.f,0.f};
    v8f acc1 = acc0;
    for (int k0 = 0; k0 < KN; k0 += 32) {
        for (int i = t; i < 16 * HD; i += 256) {
            int kp = i >> 7, c = i & 127;
            int kg = k0 + 2 * kp;
            WsT[c][kp] = pack2bf(Wn1[kg * HD + c], Wn1[(kg + 1) * HD + c]);
        }
        __syncthreads();
        AF b  = loadB(WsT, col, hf);
        AF a0 = loadA(Xs[lm],      k0, hf);
        AF a1 = loadA(Xs[16 + lm], k0, hf);
        acc0 = __builtin_amdgcn_wmma_f32_16x16x32_bf16(false, a0.v, false, b.v,
                                                       (short)0, acc0, false, false);
        acc1 = __builtin_amdgcn_wmma_f32_16x16x32_bf16(false, a1.v, false, b.v,
                                                       (short)0, acc1, false, false);
        __syncthreads();
    }

    {
        float bb = bn1[col];
        #pragma unroll
        for (int r = 0; r < 8; ++r) {
            float x = acc0[r] + bb;
            Hs[r + 8*hf][col] = f2bf(x / (1.0f + __expf(-x)));
            float y = acc1[r] + bb;
            Hs[16 + r + 8*hf][col] = f2bf(y / (1.0f + __expf(-y)));
        }
    }
    __syncthreads();

    acc0 = (v8f){0.f,0.f,0.f,0.f,0.f,0.f,0.f,0.f};
    acc1 = acc0;
    for (int k0 = 0; k0 < HD; k0 += 32) {
        for (int i = t; i < 16 * HD; i += 256) {
            int kp = i >> 7, c = i & 127;
            int kg = k0 + 2 * kp;
            WsT[c][kp] = pack2bf(Wn2[kg * ND + c], Wn2[(kg + 1) * ND + c]);
        }
        __syncthreads();
        AF b  = loadB(WsT, col, hf);
        AF a0 = loadA(Hs[lm],      k0, hf);
        AF a1 = loadA(Hs[16 + lm], k0, hf);
        acc0 = __builtin_amdgcn_wmma_f32_16x16x32_bf16(false, a0.v, false, b.v,
                                                       (short)0, acc0, false, false);
        acc1 = __builtin_amdgcn_wmma_f32_16x16x32_bf16(false, a1.v, false, b.v,
                                                       (short)0, acc1, false, false);
        __syncthreads();
    }

    {   // residual + bias, guarded f32 store
        float bb = bn2[col];
        #pragma unroll
        for (int r = 0; r < 8; ++r) {
            long row0 = n0 + r + 8*hf;
            if (row0 < N)
                out[row0 * ND + col] = nf[row0 * ND + col] + acc0[r] + bb;
            long row1 = row0 + 16;
            if (row1 < N)
                out[row1 * ND + col] = nf[row1 * ND + col] + acc1[r] + bb;
        }
    }
}

__global__ void zero_kernel(float* __restrict__ p, long n) {
    long i = (long)blockIdx.x * blockDim.x + threadIdx.x;
    long stride = (long)gridDim.x * blockDim.x;
    for (; i < n; i += stride) p[i] = 0.0f;
}

extern "C" void kernel_launch(void* const* d_in, const int* in_sizes, int n_in,
                              void* d_out, int out_size, void* d_ws, size_t ws_size,
                              hipStream_t stream)
{
    const float* nf  = (const float*)d_in[0];
    const int*   ei  = (const int*)  d_in[1];
    const float* ef  = (const float*)d_in[2];
    const float* We1 = (const float*)d_in[3];
    const float* be1 = (const float*)d_in[4];
    const float* We2 = (const float*)d_in[5];
    const float* be2 = (const float*)d_in[6];
    const float* Wi  = (const float*)d_in[7];
    const float* bi  = (const float*)d_in[8];
    const float* Wn1 = (const float*)d_in[9];
    const float* bn1 = (const float*)d_in[10];
    const float* Wn2 = (const float*)d_in[11];
    const float* bn2 = (const float*)d_in[12];

    const int N  = in_sizes[0] / ND;   // 50000
    const int nE = in_sizes[1] / 2;    // 800000

    float* agg = (float*)d_ws;         // N x 128 f32 aggregate (25.6 MB, L2-resident)
    float* out = (float*)d_out;

    zero_kernel<<<1024, 256, 0, stream>>>(agg, (long)N * MD);

    gnn_edge_kernel<<<nE / TIL, 256, 0, stream>>>(nf, ei, ef, We1, be1, We2, be2,
                                                  Wi, bi, agg, nE);
    gnn_node_kernel<<<(N + TIL - 1) / TIL, 256, 0, stream>>>(nf, agg, Wn1, bn1,
                                                             Wn2, bn2, out, N);

    // tuple tail: edge_indices (bit-copied int32) then edge_features
    char* tail = (char*)d_out + (size_t)N * ND * sizeof(float);
    hipMemcpyAsync(tail, d_in[1], (size_t)2 * nE * sizeof(int),
                   hipMemcpyDeviceToDevice, stream);
    hipMemcpyAsync(tail + (size_t)2 * nE * sizeof(int), d_in[2],
                   (size_t)nE * EDIM * sizeof(float),
                   hipMemcpyDeviceToDevice, stream);
}
